// VectorQuantizer_89421219103052
// MI455X (gfx1250) — matrix operand order repaired and checked by reading, not compiled, not run on hardware
//
#include <hip/hip_runtime.h>
#include <hip/hip_bf16.h>

typedef __attribute__((ext_vector_type(16))) _Float16 v16h;
typedef __attribute__((ext_vector_type(8)))  _Float16 v8h;
typedef __attribute__((ext_vector_type(8)))  float    v8f;

#define N_TOK   16384      // 16 * 32 * 32 tokens
#define N_E     16384      // codebook entries
#define E_DIM   256
#define BETA    0.25f
#define EMB_SCALE   16384.0f        // 2^14 (power of two: exact)
#define DOT_UNSCALE (1.0f/8192.0f)  // 2 * 2^-14
#define ROW_STRIDE  264             // halves per LDS B-tile row (256 + 8 pad)

// Map float to an unsigned key with the same total order.
__device__ __forceinline__ unsigned fkey(float f) {
  unsigned u = __float_as_uint(f);
  return (u & 0x80000000u) ? ~u : (u | 0x80000000u);
}

// Async-copy 64 contiguous bytes global -> LDS (4 x ASYNC_TO_LDS_B128).
// INST_OFFSET is added to BOTH the LDS and global addresses (ISA 15.18.3),
// so one address pair covers all four beats.  Tracked by ASYNCcnt.
__device__ __forceinline__ void stage64_async(const _Float16* gsrc, _Float16* ldst) {
  unsigned loff = (unsigned)(size_t)ldst;  // flat LDS addr truncates to LDS offset
  asm volatile(
      "global_load_async_to_lds_b128 %0, %1, off\n\t"
      "global_load_async_to_lds_b128 %0, %1, off offset:16\n\t"
      "global_load_async_to_lds_b128 %0, %1, off offset:32\n\t"
      "global_load_async_to_lds_b128 %0, %1, off offset:48"
      :: "v"(loff), "v"(gsrc) : "memory");
}
__device__ __forceinline__ void wait_async0() {
  asm volatile("s_wait_asynccnt 0x0" ::: "memory");
}

// ---------------------------------------------------------------------------
// z (b,c,h,w) f32  ->  zf16 / zf32 (n, c) with n = b*1024 + h*32 + w.
// LDS-tiled 32x32 transpose: coalesced reads AND writes.
// ---------------------------------------------------------------------------
__global__ void k_convert_z(const float* __restrict__ z,
                            _Float16* __restrict__ zf16,
                            float* __restrict__ zf32) {
  __shared__ float t[32][33];
  const int b   = blockIdx.z;
  const int hw0 = blockIdx.x * 32;
  const int c0  = blockIdx.y * 32;
  const int tx  = threadIdx.x;   // 32
  const int ty  = threadIdx.y;   // 8
#pragma unroll
  for (int i = 0; i < 4; ++i) {
    int c = ty + i * 8;
    t[c][tx] = z[(size_t)b * 262144 + (size_t)(c0 + c) * 1024 + hw0 + tx];
  }
  __syncthreads();
#pragma unroll
  for (int i = 0; i < 4; ++i) {
    int hwl = ty + i * 8;
    float v = t[tx][hwl];
    size_t n = (size_t)b * 1024 + hw0 + hwl;
    zf32[n * E_DIM + c0 + tx] = v;
    zf16[n * E_DIM + c0 + tx] = (_Float16)v;
  }
}

// ---------------------------------------------------------------------------
// emb f32 -> ef16 (scaled by 2^14 to escape f16 subnormal range) + ssq (fp32).
// ---------------------------------------------------------------------------
__global__ void k_convert_emb(const float* __restrict__ emb,
                              _Float16* __restrict__ ef16,
                              float* __restrict__ ssq) {
  __shared__ float red[256];
  const int k = blockIdx.x;
  const int c = threadIdx.x;
  float v = emb[(size_t)k * E_DIM + c];
  ef16[(size_t)k * E_DIM + c] = (_Float16)(v * EMB_SCALE);
  red[c] = v * v;
  __syncthreads();
  for (int s = 128; s > 0; s >>= 1) {
    if (c < s) red[c] += red[c + s];
    __syncthreads();
  }
  if (c == 0) ssq[k] = red[0];
}

// ---------------------------------------------------------------------------
// Per-launch init: rowmin = +inf-key, loss accumulator = 0.
// ---------------------------------------------------------------------------
__global__ void k_init(unsigned long long* __restrict__ rowmin,
                       float* __restrict__ loss_slot) {
  int i = blockIdx.x * blockDim.x + threadIdx.x;
  if (i < N_TOK) rowmin[i] = 0xFFFFFFFFFFFFFFFFull;
  if (i == 0) *loss_slot = 0.0f;
}

// ---------------------------------------------------------------------------
// Fused WMMA GEMM + argmin, double-buffered async B staging.
// Block: 256 threads = 8 waves, 128 rows; grid (128 row-blocks, 8 col-chunks).
// Each wave keeps its 16x256 f16 A tile in registers (8 x v16h).  The block
// stages 32-col x 256-K B tiles (16 KB) into one of two LDS buffers with
// GLOBAL_LOAD_ASYNC_TO_LDS_B128 while WMMAs consume the other buffer;
// s_wait_asynccnt 0 + one barrier per 16 WMMAs publishes the next tile.
// dist = |e|^2 - 2^-13 * wmma_acc; running (min,idx) per lane, merged
// cross-chunk via packed u64 global_atomic_min.
// ---------------------------------------------------------------------------
__global__ void __launch_bounds__(256) k_argmin(
    const _Float16* __restrict__ zf16,
    const _Float16* __restrict__ ef16,
    const float* __restrict__ ssq,
    unsigned long long* __restrict__ rowmin) {
  __shared__ _Float16 ldsB[2][32 * ROW_STRIDE];   // 2 x 32 rows x 256 K (+pad)

  const int lane = threadIdx.x & 31;
  const int wave = threadIdx.x >> 5;
  const int m    = lane & 15;           // A row / B col / C col within tile
  const int hi   = lane >> 4;

  const int row_base  = blockIdx.x * 128 + wave * 16;
  const int col_chunk = blockIdx.y * 2048;

  // A fragments: lane (m,hi) holds row m; K contiguous to match B:
  // elems 0..15 = K kb*32 + hi*16 .. +15.
  v16h a[8];
  {
    const _Float16* zr = zf16 + (size_t)(row_base + m) * E_DIM;
#pragma unroll
    for (int kb = 0; kb < 8; ++kb) {
      v8h lo  = *(const v8h*)(zr + kb * 32 + hi * 16);
      v8h hi8 = *(const v8h*)(zr + kb * 32 + hi * 16 + 8);
      a[kb] = __builtin_shufflevector(lo, hi8,
                0,1,2,3,4,5,6,7,8,9,10,11,12,13,14,15);
    }
  }

  float minv[8];
  int   mini[8];
#pragma unroll
  for (int r = 0; r < 8; ++r) { minv[r] = __builtin_inff(); mini[r] = 0; }

  // Staging role: thread t copies 64 contiguous bytes of emb row `trow`.
  const int trow   = threadIdx.x >> 3;  // 0..31 : B tile row
  const int tchunk = threadIdx.x & 7;   // 0..7  : 32-half chunk

  // Prologue: stage tile 0 into buffer 0.
  stage64_async(ef16 + (size_t)(col_chunk + trow) * E_DIM + tchunk * 32,
                &ldsB[0][trow * ROW_STRIDE + tchunk * 32]);
  wait_async0();
  __syncthreads();

  for (int ct = 0; ct < 64; ++ct) {
    const int cur      = ct & 1;
    const int col_base = col_chunk + ct * 32;

    // Kick async stage of the next tile into the other buffer; it flies
    // during the 16 WMMAs below.
    if (ct + 1 < 64) {
      stage64_async(ef16 + (size_t)(col_base + 32 + trow) * E_DIM + tchunk * 32,
                    &ldsB[cur ^ 1][trow * ROW_STRIDE + tchunk * 32]);
    }

    v8f acc0 = {};
    v8f acc1 = {};
#pragma unroll
    for (int kb = 0; kb < 8; ++kb) {
      // B in this code's contiguous-K order (same as A): lane (n=m, hi) holds col n, K kb*32+hi*16..+15.
      v16h b0 = *(const v16h*)(&ldsB[cur][m        * ROW_STRIDE + kb * 32 + hi * 16]);
      v16h b1 = *(const v16h*)(&ldsB[cur][(16 + m) * ROW_STRIDE + kb * 32 + hi * 16]);
      acc0 = __builtin_amdgcn_wmma_f32_16x16x32_f16(
                 false, a[kb], false, b0, (short)0, acc0, false, false);
      acc1 = __builtin_amdgcn_wmma_f32_16x16x32_f16(
                 false, a[kb], false, b1, (short)0, acc1, false, false);
    }

    const int   k0  = col_base + m;        // lane's code index, sub-tile 0
    const int   k1  = k0 + 16;             // sub-tile 1
    const float se0 = ssq[k0];
    const float se1 = ssq[k1];
#pragma unroll
    for (int r = 0; r < 8; ++r) {          // C layout: lane row = hi*8 + r
      float d0 = se0 - DOT_UNSCALE * acc0[r];
      float d1 = se1 - DOT_UNSCALE * acc1[r];
      if (d0 < minv[r]) { minv[r] = d0; mini[r] = k0; }
      if (d1 < minv[r]) { minv[r] = d1; mini[r] = k1; }
    }

    // Next tile must be resident in LDS before anyone passes the barrier.
    wait_async0();
    __syncthreads();
  }

#pragma unroll
  for (int r = 0; r < 8; ++r) {
    int grow = row_base + hi * 8 + r;
    unsigned long long packed =
        ((unsigned long long)fkey(minv[r]) << 32) | (unsigned)mini[r];
    atomicMin(&rowmin[grow], packed);   // global_atomic_min_u64
  }
}

// ---------------------------------------------------------------------------
// Gather z_q = emb[idx] back to (b,c,h,w), emit idx, accumulate loss.
// 32 tokens per block; emb rows read coalesced into LDS, written transposed.
// ---------------------------------------------------------------------------
__global__ void __launch_bounds__(256) k_gather(
    const unsigned long long* __restrict__ rowmin,
    const float* __restrict__ emb,
    const float* __restrict__ zf32,
    float* __restrict__ out_zq,
    float* __restrict__ out_loss,
    float* __restrict__ out_idx) {
  __shared__ float tile[32][257];
  __shared__ int   idx_s[32];
  __shared__ float red[256];

  const int n_base = blockIdx.x * 32;
  const int tid = threadIdx.x;

  if (tid < 32) {
    int n = n_base + tid;
    unsigned long long p = rowmin[n];
    int k = (int)(unsigned)(p & 0xFFFFFFFFull);
    idx_s[tid] = k;
    out_idx[n] = (float)k;
  }
  __syncthreads();

  float lsum = 0.0f;
  for (int r = 0; r < 32; ++r) {
    int n = n_base + r;
    int k = idx_s[r];
    float e  = emb[(size_t)k * E_DIM + tid];
    float zz = zf32[(size_t)n * E_DIM + tid];
    float dd = e - zz;
    lsum += dd * dd;
    tile[r][tid] = e;
  }

  red[tid] = lsum;
  __syncthreads();
  for (int s = 128; s > 0; s >>= 1) {
    if (tid < s) red[tid] += red[tid + s];
    __syncthreads();
  }
  if (tid == 0)
    atomicAdd(out_loss, red[0] * ((1.0f + BETA) / (float)(N_TOK * E_DIM)));

  // write (b,c,h,w): 32 consecutive hw per block stay in one b.
  const int b   = n_base >> 10;
  const int hwb = n_base & 1023;
  const int tx  = tid & 31;   // token within block (hw offset)
  const int ty  = tid >> 5;   // channel group
#pragma unroll
  for (int cc = 0; cc < 32; ++cc) {
    int c = ty * 32 + cc;
    out_zq[(size_t)b * 262144 + (size_t)c * 1024 + hwb + tx] = tile[tx][c];
  }
}

// ---------------------------------------------------------------------------
extern "C" void kernel_launch(void* const* d_in, const int* in_sizes, int n_in,
                              void* d_out, int out_size, void* d_ws, size_t ws_size,
                              hipStream_t stream) {
  const float* z   = (const float*)d_in[0];   // [16,256,32,32]
  const float* emb = (const float*)d_in[1];   // [16384,256]

  char* ws = (char*)d_ws;
  _Float16* zf16 = (_Float16*)(ws);                        //  8 MiB
  _Float16* ef16 = (_Float16*)(ws + (8u  << 20));          //  8 MiB
  float*    zf32 = (float*)   (ws + (16u << 20));          // 16 MiB
  float*    ssq  = (float*)   (ws + (32u << 20));          // 64 KiB
  unsigned long long* rowmin =
      (unsigned long long*)(ws + (32u << 20) + (64u << 10)); // 128 KiB

  float* out_zq   = (float*)d_out;            // 4,194,304 floats
  float* out_loss = out_zq + 4194304;         // 1 float
  float* out_idx  = out_loss + 1;             // 16384 "floats" (indices)

  k_convert_z  <<<dim3(32, 8, 16), dim3(32, 8), 0, stream>>>(z, zf16, zf32);
  k_convert_emb<<<dim3(N_E),       dim3(256),   0, stream>>>(emb, ef16, ssq);
  k_init       <<<dim3(64),        dim3(256),   0, stream>>>(rowmin, out_loss);
  k_argmin     <<<dim3(128, 8),    dim3(256),   0, stream>>>(zf16, ef16, ssq, rowmin);
  k_gather     <<<dim3(512),       dim3(256),   0, stream>>>(rowmin, emb, zf32,
                                                             out_zq, out_loss, out_idx);
}
